// TransformerBlock_5987184411103
// MI455X (gfx1250) — matrix operand order repaired and checked by reading, not compiled, not run on hardware
//
#include <hip/hip_runtime.h>
#include <stdint.h>

#define S_LEN 4096
#define D_DIM 512
#define BATCH 4
#define BS_TOT (BATCH * S_LEN)
#define W_ELEMS (D_DIM * D_DIM)

// Flip to 0 if the async-to-LDS mnemonic is rejected by the assembler.
#define USE_ASYNC_LDS 1

typedef __attribute__((ext_vector_type(16))) _Float16 v16h;
typedef __attribute__((ext_vector_type(8)))  _Float16 v8h;
typedef __attribute__((ext_vector_type(8)))  float    v8f;

__device__ __forceinline__ v8f wmma16(v16h a, v16h b, v8f c) {
  return __builtin_amdgcn_wmma_f32_16x16x32_f16(false, a, false, b, (short)0, c,
                                                false, false);
}

__device__ __forceinline__ float half_max16(float x) {
  x = fmaxf(x, __shfl_xor(x, 1));
  x = fmaxf(x, __shfl_xor(x, 2));
  x = fmaxf(x, __shfl_xor(x, 4));
  x = fmaxf(x, __shfl_xor(x, 8));
  return x;
}
__device__ __forceinline__ float half_sum16(float x) {
  x += __shfl_xor(x, 1); x += __shfl_xor(x, 2);
  x += __shfl_xor(x, 4); x += __shfl_xor(x, 8);
  return x;
}
__device__ __forceinline__ float wave_sum32(float x) {
  x += __shfl_xor(x, 1); x += __shfl_xor(x, 2); x += __shfl_xor(x, 4);
  x += __shfl_xor(x, 8); x += __shfl_xor(x, 16);
  return x;
}

// ---------------------------------------------------------------- kernel 0a
// xpe = x + positional_encoding (fp32) ; xpeh = f16 copy for WMMA A-fragments
__global__ __launch_bounds__(256) void pe_kernel(const float* __restrict__ x,
                                                 float* __restrict__ xpe,
                                                 _Float16* __restrict__ xpeh) {
  size_t idx = (size_t)blockIdx.x * 256 + threadIdx.x;
  int d = (int)(idx & (D_DIM - 1));
  int s = (int)((idx >> 9) & (S_LEN - 1));
  float e = (float)(1 + 2 * (d >> 1)) * (1.0f / 256.0f);
  float ang = (float)s * exp2f(-e * 13.287712379549449f); // log2(10000)
  float pe = (d & 1) ? __cosf(ang) : __sinf(ang);
  float r = x[idx] + pe;
  xpe[idx] = r;
  xpeh[idx] = (_Float16)r;
}

// ---------------------------------------------------------------- kernel 0b
// Convert Wq,Wk,Wv,Wf (fp32) into one packed f16 array Wh[4][512*512]
__global__ __launch_bounds__(256) void cvtw_kernel(
    const float* __restrict__ Wq, const float* __restrict__ Wk,
    const float* __restrict__ Wv, const float* __restrict__ Wf,
    _Float16* __restrict__ Wh) {
  size_t idx = (size_t)blockIdx.x * 256 + threadIdx.x; // 4 * 262144
  int m = (int)(idx >> 18);
  size_t off = idx & (W_ELEMS - 1);
  const float* src = (m == 0) ? Wq : (m == 1) ? Wk : (m == 2) ? Wv : Wf;
  Wh[idx] = (_Float16)src[off];
}

// ---------------------------------------------------------------- kernel 1
// Q/K (f16, row major) and V transposed Vt[b][d][s] (f16).
// W tile (64 rows x 512) staged in LDS via async-load-to-LDS, shared by 8 waves.
__global__ __launch_bounds__(256) void qkv_kernel(
    const _Float16* __restrict__ xpeh,
    const _Float16* __restrict__ Wh,
    const float* __restrict__ bq, const float* __restrict__ bk,
    const float* __restrict__ bv,
    _Float16* __restrict__ Qh, _Float16* __restrict__ Kh,
    _Float16* __restrict__ Vt) {
  __shared__ _Float16 wtile[64 * 512];   // 64 KB  B-tile
  __shared__ _Float16 tbuf[8][16 * 64];  // 16 KB  V-transpose staging
  const int lane = threadIdx.x & 31;
  const int w    = threadIdx.x >> 5;
  const int hi   = lane >> 4;
  const int col  = lane & 15;
  const int z    = blockIdx.z;
  const int row0 = blockIdx.x * 128 + w * 16;   // flat b*S+s row
  const int n0   = blockIdx.y * 64;
  const float* bias = (z == 0) ? bq : (z == 1) ? bk : bv;

  // ---- stage W[n0..n0+63][0..511] (f16) into LDS
  const char* gsrc = (const char*)(Wh + (size_t)z * W_ELEMS + (size_t)n0 * D_DIM);
#if USE_ASYNC_LDS
  {
    unsigned ldsbase = (unsigned)(uintptr_t)wtile;
#pragma unroll
    for (int i = 0; i < 16; ++i) {
      unsigned off = (threadIdx.x + 256 * i) * 16;
      asm volatile("global_load_async_to_lds_b128 %0, %1, off"
                   :: "v"(ldsbase + off), "v"(gsrc + off) : "memory");
    }
    asm volatile("s_wait_asynccnt 0" ::: "memory");
  }
#else
  {
#pragma unroll
    for (int i = 0; i < 16; ++i) {
      unsigned off = (threadIdx.x + 256 * i) * 16;
      *(v8h*)((char*)wtile + off) = *(const v8h*)(gsrc + off);
    }
  }
#endif
  __syncthreads();

  const v8f vzero = {};
  v8f c[4];
#pragma unroll
  for (int t = 0; t < 4; ++t) c[t] = vzero;

  const _Float16* xrow = xpeh + (size_t)(row0 + col) * D_DIM;
#pragma unroll 2
  for (int k0 = 0; k0 < D_DIM; k0 += 32) {
    v16h a;
    const _Float16* xr = xrow + k0 + 8 * hi;
    *(v8h*)&a       = *(const v8h*)xr;
    *((v8h*)&a + 1) = *(const v8h*)(xr + 16);
#pragma unroll
    for (int t = 0; t < 4; ++t) {
      const _Float16* wr = wtile + (size_t)(t * 16 + col) * D_DIM + k0 + 8 * hi;
      v16h bfr;
      *(v8h*)&bfr       = *(const v8h*)wr;
      *((v8h*)&bfr + 1) = *(const v8h*)(wr + 16);
      c[t] = wmma16(a, bfr, c[t]);
    }
  }

  if (z < 2) {
    _Float16* outp = (z == 0) ? Qh : Kh;
#pragma unroll
    for (int t = 0; t < 4; ++t) {
      const int n = n0 + t * 16 + col;
      const float bn = bias[n];
#pragma unroll
      for (int v = 0; v < 8; ++v) {
        const int row = row0 + v + 8 * hi;
        outp[(size_t)row * D_DIM + n] = (_Float16)(c[t][v] + bn);
      }
    }
  } else {
#pragma unroll
    for (int t = 0; t < 4; ++t) {
      const float bn = bias[n0 + t * 16 + col];
#pragma unroll
      for (int v = 0; v < 8; ++v)
        tbuf[w][(v + 8 * hi) * 64 + t * 16 + col] = (_Float16)(c[t][v] + bn);
    }
    __builtin_amdgcn_wave_barrier();
    const int b     = row0 >> 12;
    const int sbase = row0 & (S_LEN - 1);
#pragma unroll
    for (int i = 0; i < 4; ++i) {
      const int c2 = i * 32 + lane;
      const int d  = c2 >> 1;
      const int sg = c2 & 1;
      v8h pk;
#pragma unroll
      for (int r = 0; r < 8; ++r) pk[r] = tbuf[w][(sg * 8 + r) * 64 + d];
      _Float16* dst = Vt + ((size_t)b * D_DIM + n0 + d) * S_LEN + sbase + sg * 8;
      *(v8h*)dst = pk;
    }
  }
}

// ---------------------------------------------------------------- kernel 2
// Flash attention, 64-key blocks; wave owns 16 q rows x full D=512.
__global__ __launch_bounds__(128) void attn_kernel(
    const _Float16* __restrict__ Qh, const _Float16* __restrict__ Kh,
    const _Float16* __restrict__ Vt, const float* __restrict__ xpe,
    float* __restrict__ out) {
  __shared__ _Float16 pb[4][16 * 64];   // per-wave P staging (2 KB each)
  const int lane = threadIdx.x & 31;
  const int w    = threadIdx.x >> 5;
  const int hi   = lane >> 4;
  const int col  = lane & 15;
  const int b    = blockIdx.y;
  const int q0   = blockIdx.x * 64 + w * 16;
  const _Float16* Qb = Qh + ((size_t)b * S_LEN + q0) * D_DIM;
  const _Float16* Kb = Kh + (size_t)b * S_LEN * D_DIM;
  const _Float16* Vb = Vt + (size_t)b * D_DIM * S_LEN;
  _Float16* ps = pb[w];
  const float scale = 0.04419417382415922f; // 1/sqrt(512)

  float mrow[8], lrow[8];
#pragma unroll
  for (int v = 0; v < 8; ++v) { mrow[v] = -3.0e38f; lrow[v] = 0.0f; }
  const v8f vzero = {};
  v8f acc[32];
#pragma unroll
  for (int t = 0; t < 32; ++t) acc[t] = vzero;

  const _Float16* qrow = Qb + (size_t)col * D_DIM;

  for (int kb = 0; kb < S_LEN; kb += 64) {
    // ---- scores S(16x64)
    v8f s[4];
#pragma unroll
    for (int u = 0; u < 4; ++u) s[u] = vzero;
    const _Float16* kr = Kb + (size_t)(kb + col) * D_DIM + 8 * hi;
#pragma unroll 4
    for (int dk = 0; dk < D_DIM; dk += 32) {
      v16h a;
      const _Float16* qr = qrow + dk + 8 * hi;
      *(v8h*)&a       = *(const v8h*)qr;
      *((v8h*)&a + 1) = *(const v8h*)(qr + 16);
#pragma unroll
      for (int u = 0; u < 4; ++u) {
        v16h bk_;
        *(v8h*)&bk_       = *(const v8h*)(kr + (size_t)(16 * u) * D_DIM + dk);
        *((v8h*)&bk_ + 1) = *(const v8h*)(kr + (size_t)(16 * u) * D_DIM + dk + 16);
        s[u] = wmma16(a, bk_, s[u]);
      }
    }
    // ---- online softmax update
    float alpha[8];
#pragma unroll
    for (int v = 0; v < 8; ++v) {
      float x0 = s[0][v] * scale, x1 = s[1][v] * scale;
      float x2 = s[2][v] * scale, x3 = s[3][v] * scale;
      float mx = half_max16(fmaxf(fmaxf(x0, x1), fmaxf(x2, x3)));
      float mn = fmaxf(mrow[v], mx);
      float al = __expf(mrow[v] - mn);
      float p0 = __expf(x0 - mn), p1 = __expf(x1 - mn);
      float p2 = __expf(x2 - mn), p3 = __expf(x3 - mn);
      lrow[v] = lrow[v] * al + half_sum16((p0 + p1) + (p2 + p3));
      mrow[v] = mn;
      alpha[v] = al;
      _Float16* pp = ps + (v + 8 * hi) * 64 + col;
      pp[0]  = (_Float16)p0;
      pp[16] = (_Float16)p1;
      pp[32] = (_Float16)p2;
      pp[48] = (_Float16)p3;
    }
#pragma unroll
    for (int t = 0; t < 32; ++t) {
#pragma unroll
      for (int v = 0; v < 8; ++v) acc[t][v] *= alpha[v];
    }
    __builtin_amdgcn_wave_barrier();
    // ---- P(16x64) as two A-fragments from LDS
    v16h pa0, pa1;
    const _Float16* pr = ps + col * 64 + 8 * hi;
    *(v8h*)&pa0       = *(const v8h*)pr;
    *((v8h*)&pa0 + 1) = *(const v8h*)(pr + 16);
    *(v8h*)&pa1       = *(const v8h*)(pr + 32);
    *((v8h*)&pa1 + 1) = *(const v8h*)(pr + 48);
    // ---- acc += P @ V (two passes so adjacent WMMAs hit different acc tiles)
    const _Float16* vcol = Vb + (size_t)col * S_LEN + kb + 8 * hi;
#pragma unroll
    for (int t = 0; t < 32; ++t) {
      v16h bv;
      *(v8h*)&bv       = *(const v8h*)(vcol + (size_t)t * 16 * S_LEN);
      *((v8h*)&bv + 1) = *(const v8h*)(vcol + 16 + (size_t)t * 16 * S_LEN);
      acc[t] = wmma16(pa0, bv, acc[t]);
    }
#pragma unroll
    for (int t = 0; t < 32; ++t) {
      v16h bv;
      *(v8h*)&bv       = *(const v8h*)(vcol + 32 + (size_t)t * 16 * S_LEN);
      *((v8h*)&bv + 1) = *(const v8h*)(vcol + 48 + (size_t)t * 16 * S_LEN);
      acc[t] = wmma16(pa1, bv, acc[t]);
    }
  }
  float invl[8];
#pragma unroll
  for (int v = 0; v < 8; ++v) invl[v] = 1.0f / lrow[v];
#pragma unroll
  for (int t = 0; t < 32; ++t) {
    const int d = t * 16 + col;
#pragma unroll
    for (int v = 0; v < 8; ++v) {
      const size_t idx = ((size_t)b * S_LEN + q0 + v + 8 * hi) * D_DIM + d;
      out[idx] = xpe[idx] + acc[t][v] * invl[v];
    }
  }
}

// ---------------------------------------------------------------- kernel 3
// x = LN(x); out = x + x@Wf^T + bf   (in-place on d_out; block owns its rows)
__global__ __launch_bounds__(128) void lnff_kernel(
    const _Float16* __restrict__ Whf, const float* __restrict__ bf,
    const float* __restrict__ ln_g, const float* __restrict__ ln_b,
    float* __restrict__ out) {
  __shared__ _Float16 xn[64 * 512];   // 64 KB normalized rows (f16)
  const int lane = threadIdx.x & 31;
  const int w    = threadIdx.x >> 5;
  const int hi   = lane >> 4;
  const int col  = lane & 15;
  const int row0 = blockIdx.x * 64;

  for (int r = 0; r < 16; ++r) {
    const int m = w * 16 + r;
    const size_t base = (size_t)(row0 + m) * D_DIM;
    float vals[16];
    float s = 0.0f, ss = 0.0f;
#pragma unroll
    for (int i = 0; i < 16; ++i) {
      float x = out[base + lane + 32 * i];
      vals[i] = x; s += x; ss += x * x;
    }
    s = wave_sum32(s); ss = wave_sum32(ss);
    const float mu = s * (1.0f / 512.0f);
    const float rstd = rsqrtf(ss * (1.0f / 512.0f) - mu * mu + 1e-5f);
#pragma unroll
    for (int i = 0; i < 16; ++i) {
      const int d = lane + 32 * i;
      xn[m * 512 + d] = (_Float16)((vals[i] - mu) * rstd * ln_g[d] + ln_b[d]);
    }
  }
  __syncthreads();

  for (int nb = 0; nb < 4; ++nb) {
    const int n0 = nb * 128;
    const v8f vzero = {};
    v8f c[8];
#pragma unroll
    for (int t = 0; t < 8; ++t) c[t] = vzero;
    const _Float16* arow = xn + (w * 16 + col) * 512;
#pragma unroll 2
    for (int k0 = 0; k0 < D_DIM; k0 += 32) {
      v16h a;
      const _Float16* ar = arow + k0 + 8 * hi;
      *(v8h*)&a       = *(const v8h*)ar;
      *((v8h*)&a + 1) = *(const v8h*)(ar + 16);
#pragma unroll
      for (int t = 0; t < 8; ++t) {
        const _Float16* wr = Whf + (size_t)(n0 + t * 16 + col) * D_DIM + k0 + 8 * hi;
        v16h bfr;
        *(v8h*)&bfr       = *(const v8h*)wr;
        *((v8h*)&bfr + 1) = *(const v8h*)(wr + 16);
        c[t] = wmma16(a, bfr, c[t]);
      }
    }
#pragma unroll
    for (int t = 0; t < 8; ++t) {
      const int n = n0 + t * 16 + col;
      const float bn = bf[n];
#pragma unroll
      for (int v = 0; v < 8; ++v) {
        const int m = w * 16 + v + 8 * hi;
        out[(size_t)(row0 + m) * D_DIM + n] = c[t][v] + bn + (float)xn[m * 512 + n];
      }
    }
  }
}

// ---------------------------------------------------------------- launcher
extern "C" void kernel_launch(void* const* d_in, const int* in_sizes, int n_in,
                              void* d_out, int out_size, void* d_ws, size_t ws_size,
                              hipStream_t stream) {
  (void)in_sizes; (void)n_in; (void)out_size; (void)ws_size;
  const float* x    = (const float*)d_in[0];
  const float* Wq   = (const float*)d_in[1];
  const float* bq   = (const float*)d_in[2];
  const float* Wk   = (const float*)d_in[3];
  const float* bk   = (const float*)d_in[4];
  const float* Wv   = (const float*)d_in[5];
  const float* bv   = (const float*)d_in[6];
  const float* ln_g = (const float*)d_in[7];
  const float* ln_b = (const float*)d_in[8];
  const float* Wf   = (const float*)d_in[9];
  const float* bf   = (const float*)d_in[10];
  float* out = (float*)d_out;

  char* wsb = (char*)d_ws;
  float*    xpe  = (float*)wsb;                                   // 32 MB
  _Float16* Qh   = (_Float16*)(wsb + (size_t)BS_TOT * D_DIM * 4); // 16 MB
  _Float16* Kh   = Qh + (size_t)BS_TOT * D_DIM;                   // 16 MB
  _Float16* Vt   = Kh + (size_t)BS_TOT * D_DIM;                   // 16 MB
  _Float16* xpeh = Vt + (size_t)BS_TOT * D_DIM;                   // 16 MB
  _Float16* Wh   = xpeh + (size_t)BS_TOT * D_DIM;                 //  2 MB (4 mats)

  pe_kernel<<<(BS_TOT * D_DIM) / 256, 256, 0, stream>>>(x, xpe, xpeh);
  cvtw_kernel<<<(4 * W_ELEMS) / 256, 256, 0, stream>>>(Wq, Wk, Wv, Wf, Wh);
  qkv_kernel<<<dim3(BS_TOT / 128, D_DIM / 64, 3), 256, 0, stream>>>(
      xpeh, Wh, bq, bk, bv, Qh, Kh, Vt);
  attn_kernel<<<dim3(S_LEN / 64, BATCH), 128, 0, stream>>>(Qh, Kh, Vt, xpe, out);
  lnff_kernel<<<BS_TOT / 64, 128, 0, stream>>>(Wh + (size_t)3 * W_ELEMS, bf,
                                               ln_g, ln_b, out);
}
